// PQHotLowRank_89593017794940
// MI455X (gfx1250) — matrix operand order repaired and checked by reading, not compiled, not run on hardware
//
#include <hip/hip_runtime.h>

typedef float v2f  __attribute__((ext_vector_type(2)));
typedef float v8f  __attribute__((ext_vector_type(8)));
typedef __bf16 v16bf __attribute__((ext_vector_type(16)));

#define K_ROWS 32768   // rows of U / output
#define R_DIM  256     // inner dim (cols of U, rows of B)
#define D_COLS 1024    // cols of B / output
#define KCB    256     // codebook entries
#define DGRP   8       // PQ group size

// ---- bf16 split helpers (RNE) ----
static __device__ __forceinline__ unsigned short f2bf(float f) {
  unsigned u = __float_as_uint(f);
  u += 0x7FFFu + ((u >> 16) & 1u);
  return (unsigned short)(u >> 16);
}
static __device__ __forceinline__ float bf2f(unsigned short b) {
  return __uint_as_float(((unsigned)b) << 16);
}

// ======================= Phase 1: quantize B (small) =======================
// One thread per group (32768 groups). Emits Bq as a bf16 hi/lo split,
// pre-swizzled into this file's B-fragment order for V_WMMA_F32_16X16X32_BF16 (K contiguous per lane half):
//   u32[((kc*64 + nt)*32 + lane)*8 + j] = pair (K = kc*32+16*(lane>>4)+2j, +1), col = nt*16+(lane&15)
__global__ __launch_bounds__(256) void pq_quant_B(const float* __restrict__ B,
                                                  const float* __restrict__ rsB,
                                                  const float* __restrict__ cbB,
                                                  unsigned int* __restrict__ Bwh,
                                                  unsigned int* __restrict__ Bwl) {
  __shared__ float cbs[KCB * DGRP];
  __shared__ float hns[KCB];
  const int tid = threadIdx.x;
  float4 c0 = ((const float4*)cbB)[tid * 2 + 0];
  float4 c1 = ((const float4*)cbB)[tid * 2 + 1];
  ((float4*)cbs)[tid * 2 + 0] = c0;
  ((float4*)cbs)[tid * 2 + 1] = c1;
  hns[tid] = 0.5f * (c0.x * c0.x + c0.y * c0.y + c0.z * c0.z + c0.w * c0.w +
                     c1.x * c1.x + c1.y * c1.y + c1.z * c1.z + c1.w * c1.w);
  __syncthreads();

  const int gi   = blockIdx.x * 256 + tid;      // group id in [0, 32768)
  const int r    = gi >> 7;                     // K index (row of B), 128 groups/row
  const int col0 = (gi & 127) * DGRP;
  const float rs   = rsB[r];
  const float invs = 1.0f / rs;

  float x[8];
  float4 x0 = *(const float4*)(B + (size_t)r * D_COLS + col0);
  float4 x1 = *(const float4*)(B + (size_t)r * D_COLS + col0 + 4);
  x[0] = x0.x * invs; x[1] = x0.y * invs; x[2] = x0.z * invs; x[3] = x0.w * invs;
  x[4] = x1.x * invs; x[5] = x1.y * invs; x[6] = x1.z * invs; x[7] = x1.w * invs;

  float best = 1e30f;
  int   bidx = 0;
  for (int c = 0; c < KCB; ++c) {
    float s = hns[c];
    const float* cp = &cbs[c * DGRP];
    #pragma unroll
    for (int j = 0; j < 8; ++j) s -= x[j] * cp[j];
    bool lt = s < best;            // strict < => first index on ties (matches argmin)
    bidx = lt ? c : bidx;
    best = lt ? s : best;
  }
  const float* qp = &cbs[bidx * DGRP];

  // scatter hi/lo bf16 into the swizzled B-fragment layout
  const int kc  = r >> 5;
  const int rr  = r & 31;
  const int hh  = rr >> 4;
  const int jj  = (rr & 15) >> 1;
  const int par = rr & 1;
  #pragma unroll
  for (int j = 0; j < 8; ++j) {
    const int c  = col0 + j;
    const int nt = c >> 4;
    const int ln = (hh << 4) | (c & 15);
    const size_t u32idx = ((size_t)(kc * 64 + nt) * 32 + ln) * 8 + jj;
    const size_t u16idx = u32idx * 2 + par;
    float v = qp[j] * rs;
    unsigned short hb = f2bf(v);
    unsigned short lb = f2bf(v - bf2f(hb));
    ((unsigned short*)Bwh)[u16idx] = hb;
    ((unsigned short*)Bwl)[u16idx] = lb;
  }
}

// ======================= Phase 2: quantize U via WMMA =======================
// Distance argmin as fp32 WMMA with K extended 8 -> 12 (half-norm bias rides
// in the matrix op). Packed-double min tracks value+index with no masks.
// Epilogue emits Uq as row-major bf16 hi/lo split for the bf16-split GEMM.
__global__ __launch_bounds__(256) void pq_quant_U(const float* __restrict__ U,
                                                  const float* __restrict__ rsU,
                                                  const float* __restrict__ cbU,
                                                  unsigned short* __restrict__ Uqh,
                                                  unsigned short* __restrict__ Uql) {
  const int lane = threadIdx.x & 31;
  const int lm   = lane & 15;
  const int h    = lane >> 4;
  const int wid  = (blockIdx.x * blockDim.x + threadIdx.x) >> 5;
  const int nwav = (gridDim.x * blockDim.x) >> 5;

  // Preload codebook^T B-fragments (16 code tiles x 2 k-steps) and half-norms.
  v2f   bf[16][2];
  float hn[16];
  #pragma unroll
  for (int t = 0; t < 16; ++t) {
    const float* cp = cbU + (t * 16 + lm) * DGRP;
    float4 q0 = *(const float4*)(cp);
    float4 q1 = *(const float4*)(cp + 4);
    bf[t][0].x = h ? q0.z : q0.x;  bf[t][0].y = h ? q0.w : q0.y;
    bf[t][1].x = h ? q1.z : q1.x;  bf[t][1].y = h ? q1.w : q1.y;
    hn[t] = 0.5f * (q0.x * q0.x + q0.y * q0.y + q0.z * q0.z + q0.w * q0.w +
                    q1.x * q1.x + q1.y * q1.y + q1.z * q1.z + q1.w * q1.w);
  }
  // Constant A-fragment for the bias k-step (K rows 8..11): A[m][8] = 1, rest 0.
  v2f a2;
  a2.x = h ? 0.0f : 1.0f;
  a2.y = 0.0f;

  for (int row = wid; row < K_ROWS; row += nwav) {
    const float rs    = rsU[row];
    const float ninvs = -1.0f / rs;             // negate A at load (f32 WMMA has no A-neg)
    const float* urow = U + (size_t)row * R_DIM;

    #pragma unroll
    for (int blk = 0; blk < 2; ++blk) {         // 2 blocks of 16 groups per row
      const float* ub = urow + blk * 128;
      v2f a0 = *(const v2f*)(ub + lm * 8 + 2 * h);
      v2f a1 = *(const v2f*)(ub + lm * 8 + 4 + 2 * h);
      a0.x *= ninvs; a0.y *= ninvs;
      a1.x *= ninvs; a1.y *= ninvs;

      double pb[8];
      #pragma unroll
      for (int i = 0; i < 8; ++i) pb[i] = 1e300;

      #pragma unroll
      for (int t = 0; t < 16; ++t) {
        v2f b2;                                 // B[8][code] = hn[code], rows 9..11 = 0
        b2.x = h ? 0.0f : hn[t];
        b2.y = 0.0f;
        v8f z = {};
        v8f c;
        c = __builtin_amdgcn_wmma_f32_16x16x4_f32(false, a2, false, b2,
                                                  (short)0, z, false, false);
        c = __builtin_amdgcn_wmma_f32_16x16x4_f32(false, a0, false, bf[t][0],
                                                  (short)0, c, false, false);
        c = __builtin_amdgcn_wmma_f32_16x16x4_f32(false, a1, false, bf[t][1],
                                                  (short)0, c, false, false);
        const int cand = t * 16 + lm;           // 8-bit code of this lane's column
        #pragma unroll
        for (int i = 0; i < 8; ++i) {
          double ps = (double)c[i];             // low 29 mantissa bits are zero
          double pd = __hiloint2double(__double2hiint(ps),
                                       __double2loint(ps) | cand);
          pb[i] = fmin(pb[i], pd);
        }
      }

      // argmin across the 16 lanes of each half-wave (pure min_f64 butterfly)
      #pragma unroll
      for (int i = 0; i < 8; ++i) {
        #pragma unroll
        for (int off = 8; off >= 1; off >>= 1) {
          int ohi = __shfl_xor(__double2hiint(pb[i]), off, 16);
          int olo = __shfl_xor(__double2loint(pb[i]), off, 16);
          pb[i] = fmin(pb[i], __hiloint2double(ohi, olo));
        }
      }

      // Lane L writes elements f = 4L..4L+3 of the 128-elt block.
      const int s = (lane >> 1) & 7;
      int code = __double2loint(pb[0]) & 0xFF;
      #pragma unroll
      for (int i = 1; i < 8; ++i) {
        int ci = __double2loint(pb[i]) & 0xFF;
        code = (s == i) ? ci : code;
      }

      float4 q = *(const float4*)(cbU + code * DGRP + (lane & 1) * 4);
      float v0 = q.x * rs, v1 = q.y * rs, v2 = q.z * rs, v3 = q.w * rs;
      unsigned short h0 = f2bf(v0), h1 = f2bf(v1), h2 = f2bf(v2), h3 = f2bf(v3);
      uint2 ph, pl;
      ph.x = (unsigned)h0 | ((unsigned)h1 << 16);
      ph.y = (unsigned)h2 | ((unsigned)h3 << 16);
      pl.x = (unsigned)f2bf(v0 - bf2f(h0)) | ((unsigned)f2bf(v1 - bf2f(h1)) << 16);
      pl.y = (unsigned)f2bf(v2 - bf2f(h2)) | ((unsigned)f2bf(v3 - bf2f(h3)) << 16);
      const size_t off16 = (size_t)row * R_DIM + blk * 128 + lane * 4;
      *(uint2*)(Uqh + off16) = ph;
      *(uint2*)(Uql + off16) = pl;
    }
  }
}

// ================= Phase 3: Uq @ Bq via bf16-split WMMA =================
// A*B ~= Ah*Bh + Ah*Bl + Al*Bh using V_WMMA_F32_16X16X32_BF16 (fp32 accum).
// 24 matrix ops per 16x16 tile instead of 64 fp32 x4 ops; error ~2^-17.
static __device__ __forceinline__ v16bf ld_a_frag(const unsigned short* p) {
  union { uint4 q[2]; v16bf v; } u;
  u.q[0] = *(const uint4*)(p);        // K run 0: 8 contiguous bf16
  u.q[1] = *(const uint4*)(p + 8);    // K run 1: +8 elements
  return u.v;
}
static __device__ __forceinline__ v16bf ld_b_frag(const unsigned int* p) {
  union { uint4 q[2]; v16bf v; } u;
  u.q[0] = *(const uint4*)(p);        // pre-swizzled: 8 contiguous dwords
  u.q[1] = *(const uint4*)(p + 4);
  return u.v;
}

__global__ __launch_bounds__(256) void pq_gemm(const unsigned short* __restrict__ Uqh,
                                               const unsigned short* __restrict__ Uql,
                                               const unsigned int* __restrict__ Bwh,
                                               const unsigned int* __restrict__ Bwl,
                                               float* __restrict__ out) {
  const int lane = threadIdx.x & 31;
  const int lm   = lane & 15;
  const int h    = lane >> 4;
  const int wid  = (blockIdx.x * blockDim.x + threadIdx.x) >> 5;
  const int nwav = (gridDim.x * blockDim.x) >> 5;
  const int NT   = D_COLS / 64;                 // 16 N-jobs per M-row (64 cols each)
  const int JOBS = (K_ROWS / 32) * NT;          // 16384 wave-jobs

  for (int job = wid; job < JOBS; job += nwav) {
    const int m0  = (job / NT) * 32;
    const int nt0 = (job % NT) * 4;             // first of 4 16-col subtiles

    v8f acc[2][4];
    #pragma unroll
    for (int s2 = 0; s2 < 2; ++s2)
      #pragma unroll
      for (int nt = 0; nt < 4; ++nt)
        #pragma unroll
        for (int i = 0; i < 8; ++i) acc[s2][nt][i] = 0.0f;

    #pragma unroll 2
    for (int kc = 0; kc < 8; ++kc) {            // K chunks of 32
      v16bf ah[2], al[2];
      #pragma unroll
      for (int s2 = 0; s2 < 2; ++s2) {
        const size_t ro = (size_t)(m0 + s2 * 16 + lm) * R_DIM + kc * 32 + 16 * h;
        ah[s2] = ld_a_frag(Uqh + ro);
        al[s2] = ld_a_frag(Uql + ro);
      }
      #pragma unroll
      for (int nt = 0; nt < 4; ++nt) {
        const size_t bo = ((size_t)(kc * 64 + nt0 + nt) * 32 + lane) * 8;
        v16bf bh = ld_b_frag(Bwh + bo);
        v16bf bl = ld_b_frag(Bwl + bo);
        #pragma unroll
        for (int s2 = 0; s2 < 2; ++s2) {
          acc[s2][nt] = __builtin_amdgcn_wmma_f32_16x16x32_bf16(false, al[s2], false, bh,
                                                                (short)0, acc[s2][nt], false, false);
          acc[s2][nt] = __builtin_amdgcn_wmma_f32_16x16x32_bf16(false, ah[s2], false, bl,
                                                                (short)0, acc[s2][nt], false, false);
          acc[s2][nt] = __builtin_amdgcn_wmma_f32_16x16x32_bf16(false, ah[s2], false, bh,
                                                                (short)0, acc[s2][nt], false, false);
        }
      }
    }

    // Output is write-once (128 MB): non-temporal stores keep Uq/Bq hot in L2.
    #pragma unroll
    for (int s2 = 0; s2 < 2; ++s2)
      #pragma unroll
      for (int nt = 0; nt < 4; ++nt)
        #pragma unroll
        for (int i = 0; i < 8; ++i)
          __builtin_nontemporal_store(
              acc[s2][nt][i],
              out + (size_t)(m0 + s2 * 16 + 8 * h + i) * D_COLS + (nt0 + nt) * 16 + lm);
  }
}

// ================================ launcher ================================
extern "C" void kernel_launch(void* const* d_in, const int* in_sizes, int n_in,
                              void* d_out, int out_size, void* d_ws, size_t ws_size,
                              hipStream_t stream) {
  const float* U    = (const float*)d_in[0];   // [32768, 256]
  const float* B    = (const float*)d_in[1];   // [256, 1024]
  const float* rsU  = (const float*)d_in[2];   // [32768, 1]
  const float* rsB  = (const float*)d_in[3];   // [256, 1]
  const float* cbU  = (const float*)d_in[4];   // [256, 8]
  const float* cbB  = (const float*)d_in[5];   // [256, 8]
  // d_in[6] is D == 8 (compile-time constant here)

  unsigned short* Uqh = (unsigned short*)d_ws;                    // 16 MB
  unsigned short* Uql = Uqh + (size_t)K_ROWS * R_DIM;             // 16 MB
  unsigned int*   Bwh = (unsigned int*)(Uql + (size_t)K_ROWS * R_DIM); // 512 KB
  unsigned int*   Bwl = Bwh + (R_DIM * D_COLS / 2);               // 512 KB
  float* out = (float*)d_out;

  // Phase 1: quantize B + emit swizzled bf16 hi/lo fragments
  pq_quant_B<<<(R_DIM * D_COLS / DGRP) / 256, 256, 0, stream>>>(B, rsB, cbB, Bwh, Bwl);
  // Phase 2: quantize U (WMMA distance search) + emit bf16 hi/lo split
  pq_quant_U<<<1024, 256, 0, stream>>>(U, rsU, cbU, Uqh, Uql);
  // Phase 3: bf16-split GEMM (one wave per 32x64 tile; 16384 jobs)
  pq_gemm<<<2048, 256, 0, stream>>>(Uqh, Uql, Bwh, Bwl, out);
}
